// SpikingKAN_14405320311288
// MI455X (gfx1250) — matrix operand order repaired and checked by reading, not compiled, not run on hardware
//
#include <hip/hip_runtime.h>
#include <hip/hip_bf16.h>
#include <math.h>

typedef _Float16 v8h  __attribute__((ext_vector_type(8)));
typedef _Float16 v16h __attribute__((ext_vector_type(16)));
typedef float    v8f  __attribute__((ext_vector_type(8)));

#define B_SZ   1024
#define IN_SZ  1024
#define H1_SZ  2048
#define H2_SZ  1024
#define NC_SZ  1000
#define NC_PAD 1024
#define DEG    8
#define K1     (IN_SZ * (DEG + 1))   // 9216

// ---------------------------------------------------------------------------
// Elementwise: Chebyshev basis of tanh(0.5 x) -> f16 A-matrix [B][I*9]
// ---------------------------------------------------------------------------
__global__ __launch_bounds__(256)
void skan_basis_f16(const float* __restrict__ x, _Float16* __restrict__ A,
                    long n /* = B*I */, int I) {
    long idx = (long)blockIdx.x * blockDim.x + threadIdx.x;
    long stride = (long)gridDim.x * blockDim.x;
    for (; idx < n; idx += stride) {
        float xn = tanhf(0.5f * x[idx]);
        _Float16* p = A + idx * (DEG + 1);
        float tm2 = 1.0f, tm1 = xn;
        p[0] = (_Float16)1.0f;
        p[1] = (_Float16)xn;
        #pragma unroll
        for (int d = 2; d <= DEG; ++d) {
            float t = 2.0f * xn * tm1 - tm2;
            p[d] = (_Float16)t;
            tm2 = tm1; tm1 = t;
        }
    }
}

// ---------------------------------------------------------------------------
// Elementwise f32 -> f16 convert (coeffs1 is already [N][K] = [2048][9216])
// ---------------------------------------------------------------------------
__global__ __launch_bounds__(256)
void skan_cvt_f16(const float* __restrict__ in, _Float16* __restrict__ out, long n) {
    long idx = (long)blockIdx.x * blockDim.x + threadIdx.x;
    long stride = (long)gridDim.x * blockDim.x;
    for (; idx < n; idx += stride) out[idx] = (_Float16)in[idx];
}

// w [1000][1024] -> f16 [1024][1024], zero pad rows >= 1000
__global__ __launch_bounds__(256)
void skan_cvt_w_pad(const float* __restrict__ w, _Float16* __restrict__ out,
                    int nreal, int npad, int k) {
    long n = (long)npad * k;
    long idx = (long)blockIdx.x * blockDim.x + threadIdx.x;
    long stride = (long)gridDim.x * blockDim.x;
    for (; idx < n; idx += stride) {
        int row = (int)(idx / k);
        out[idx] = (row < nreal) ? (_Float16)w[idx] : (_Float16)0.0f;
    }
}

// ---------------------------------------------------------------------------
// Collapse coeffs2 over degree axis for binary spike input:
//   c0[o,i] = sum_d T_d(0)    * c2[o,i,d]
//   c1[o,i] = sum_d T_d(th05) * c2[o,i,d],  th05 = tanh(0.5)
//   dc[o][i] = c1-c0 (f16 weight, [N][K] layout);  base[o] = sum_i c0
// ---------------------------------------------------------------------------
__global__ __launch_bounds__(256)
void skan_make_dc(const float* __restrict__ c2, _Float16* __restrict__ dc,
                  float* __restrict__ base, int I) {
    const int o = blockIdx.x;
    const float t = 0.46211715726000974f;  // tanh(0.5)
    float T[DEG + 1];
    T[0] = 1.0f; T[1] = t;
    #pragma unroll
    for (int d = 2; d <= DEG; ++d) T[d] = 2.0f * t * T[d - 1] - T[d - 2];

    float local = 0.0f;
    for (int i = threadIdx.x; i < I; i += 256) {
        const float* c = c2 + ((long)o * I + i) * (DEG + 1);
        float c0 = c[0] - c[2] + c[4] - c[6] + c[8];   // T_d(0) = 1,0,-1,0,1,0,-1,0,1
        float c1 = 0.0f;
        #pragma unroll
        for (int d = 0; d <= DEG; ++d) c1 += T[d] * c[d];
        dc[(long)o * I + i] = (_Float16)(c1 - c0);
        local += c0;
    }
    __shared__ float red[256];
    red[threadIdx.x] = local;
    __syncthreads();
    for (int s = 128; s > 0; s >>= 1) {
        if (threadIdx.x < s) red[threadIdx.x] += red[threadIdx.x + s];
        __syncthreads();
    }
    if (threadIdx.x == 0) base[o] = red[0];
}

// ---------------------------------------------------------------------------
// WMMA GEMM:  C[m][n] = (sum_k A[m][k]*Bt[n][k] + colAdd[n]) * colScale[n] + colBias[n]
// A:[M][K] f16 row-major, Bt:[Npad][K] f16 row-major. M%16==0, Npad%64==0, K%32==0.
// 8 waves / block, each wave owns a 16x64 macro-tile (4 accumulators, A reused 4x).
// ---------------------------------------------------------------------------
__global__ __launch_bounds__(256)
void skan_gemm_f16_wmma(const _Float16* __restrict__ A,
                        const _Float16* __restrict__ Bt,
                        float* __restrict__ C,
                        const float* __restrict__ colAdd,
                        const float* __restrict__ colScale,
                        const float* __restrict__ colBias,
                        int M, int Npad, int Nout, int K) {
    const int wave = threadIdx.x >> 5;
    const int lane = threadIdx.x & 31;
    const int tilesN4 = Npad >> 6;
    const int wid = blockIdx.x * 8 + wave;
    const int tm  = wid / tilesN4;
    const int tn4 = wid % tilesN4;
    if (tm * 16 >= M) return;

    // A fragment: row = lane&15; elems 0..15 = K[kb..kb+15] (contiguous)
    const int arow = lane & 15;
    const int kg   = (lane >> 4) << 3;                // 0 or 8
    // B fragment: col = lane&15; elems 0..15 = K[kb..kb+15] (contiguous)
    const int kb   = (lane >> 4) << 4;                // 0 or 16
    const _Float16* Ap = A  + (long)(tm * 16 + arow) * K;
    const _Float16* Bp = Bt + (long)(tn4 * 64 + arow) * K;

    v8f acc[4] = {v8f{}, v8f{}, v8f{}, v8f{}};

    for (int k0 = 0; k0 < K; k0 += 32) {
        v16h a = *(const v16h*)(Ap + k0 + kb);
        #pragma unroll
        for (int j = 0; j < 4; ++j) {
            v16h b = *(const v16h*)(Bp + (long)j * 16 * K + k0 + kb);
            acc[j] = __builtin_amdgcn_wmma_f32_16x16x32_f16(
                         false, a, false, b, (short)0, acc[j], false, false);
        }
    }

    // C/D layout: n = lane&15; vgpr r -> m = (lane>>4)*8 + r
    const int ncol  = lane & 15;
    const int mbase = (lane >> 4) << 3;
    #pragma unroll
    for (int j = 0; j < 4; ++j) {
        const int n = tn4 * 64 + j * 16 + ncol;
        if (n >= Nout) continue;
        const float add = colAdd   ? colAdd[n]   : 0.0f;
        const float scl = colScale ? colScale[n] : 1.0f;
        const float bia = colBias  ? colBias[n]  : 0.0f;
        #pragma unroll
        for (int r = 0; r < 8; ++r) {
            const int m = tm * 16 + mbase + r;
            C[(long)m * Nout + n] = (acc[j][r] + add) * scl + bia;
        }
    }
}

// ---------------------------------------------------------------------------
// Per-column batch stats (training-mode BN, biased variance)
// ---------------------------------------------------------------------------
__global__ __launch_bounds__(256)
void skan_bn_stats(const float* __restrict__ h, float* __restrict__ mean,
                   float* __restrict__ var, int Brows, int O) {
    const int o = blockIdx.x;
    float s = 0.0f, s2 = 0.0f;
    for (int r = threadIdx.x; r < Brows; r += 256) {
        float v = h[(long)r * O + o];
        s += v; s2 += v * v;
    }
    __shared__ float rs[256], rs2[256];
    rs[threadIdx.x] = s; rs2[threadIdx.x] = s2;
    __syncthreads();
    for (int st = 128; st > 0; st >>= 1) {
        if (threadIdx.x < st) {
            rs[threadIdx.x]  += rs[threadIdx.x + st];
            rs2[threadIdx.x] += rs2[threadIdx.x + st];
        }
        __syncthreads();
    }
    if (threadIdx.x == 0) {
        float mu = rs[0] / (float)Brows;
        mean[o] = mu;
        var[o]  = rs2[0] / (float)Brows - mu * mu;
    }
}

// ---------------------------------------------------------------------------
// Fused BN + LIF spike; emits fp32 (output slice) and f16 (next GEMM A)
// ---------------------------------------------------------------------------
__global__ __launch_bounds__(256)
void skan_bn_spike(const float* __restrict__ h, const float* __restrict__ mean,
                   const float* __restrict__ var, const float* __restrict__ gamma,
                   const float* __restrict__ beta, float* __restrict__ s_f32,
                   _Float16* __restrict__ s_f16, long n, int O) {
    long idx = (long)blockIdx.x * blockDim.x + threadIdx.x;
    long stride = (long)gridDim.x * blockDim.x;
    for (; idx < n; idx += stride) {
        int o = (int)(idx % O);
        float v = gamma[o] * (h[idx] - mean[o]) * rsqrtf(var[o] + 1e-5f) + beta[o];
        float sp = (0.5f * v >= 1.0f) ? 1.0f : 0.0f;   // mem = h/tau, tau=2
        s_f32[idx] = sp;
        s_f16[idx] = (_Float16)sp;
    }
}

// ---------------------------------------------------------------------------
// Launcher
// ---------------------------------------------------------------------------
extern "C" void kernel_launch(void* const* d_in, const int* in_sizes, int n_in,
                              void* d_out, int out_size, void* d_ws, size_t ws_size,
                              hipStream_t stream) {
    const float* x       = (const float*)d_in[0];
    const float* coeffs1 = (const float*)d_in[1];
    const float* scale1  = (const float*)d_in[2];
    const float* gamma1  = (const float*)d_in[3];
    const float* beta1   = (const float*)d_in[4];
    const float* coeffs2 = (const float*)d_in[5];
    const float* scale2  = (const float*)d_in[6];
    const float* gamma2  = (const float*)d_in[7];
    const float* beta2   = (const float*)d_in[8];
    const float* w       = (const float*)d_in[9];
    const float* bvec    = (const float*)d_in[10];

    float* out   = (float*)d_out;                       // [1024][1000]
    float* s1out = out + (long)B_SZ * NC_SZ;            // [1024][2048]
    float* s2out = s1out + (long)B_SZ * H1_SZ;          // [1024][1024]

    // workspace bump allocator (256B aligned)
    char* ws = (char*)d_ws;
    size_t off = 0;
    auto alloc = [&](size_t bytes) -> char* {
        char* p = ws + off;
        off += (bytes + 255) & ~(size_t)255;
        return p;
    };
    _Float16* A1   = (_Float16*)alloc((size_t)B_SZ * K1 * 2);        // basis
    _Float16* Bt1  = (_Float16*)alloc((size_t)H1_SZ * K1 * 2);       // coeffs1 f16
    float*    h    = (float*)   alloc((size_t)B_SZ * H1_SZ * 4);     // h1 / h2 (reused)
    _Float16* dc   = (_Float16*)alloc((size_t)H2_SZ * H1_SZ * 2);    // layer2 weights
    float*    base = (float*)   alloc((size_t)H2_SZ * 4);
    _Float16* A2   = (_Float16*)alloc((size_t)B_SZ * H1_SZ * 2);     // s1 f16
    _Float16* A3   = (_Float16*)alloc((size_t)B_SZ * H2_SZ * 2);     // s2 f16
    _Float16* wt   = (_Float16*)alloc((size_t)NC_PAD * H2_SZ * 2);   // padded w f16
    float*    mu   = (float*)   alloc((size_t)H1_SZ * 4);
    float*    vr   = (float*)   alloc((size_t)H1_SZ * 4);
    (void)ws_size; (void)n_in; (void)in_sizes; (void)out_size;

    // --- prep ---
    skan_cvt_f16<<<1024, 256, 0, stream>>>(coeffs1, Bt1, (long)H1_SZ * K1);
    skan_cvt_w_pad<<<512, 256, 0, stream>>>(w, wt, NC_SZ, NC_PAD, H2_SZ);
    skan_make_dc<<<H2_SZ, 256, 0, stream>>>(coeffs2, dc, base, H1_SZ);
    skan_basis_f16<<<1024, 256, 0, stream>>>(x, A1, (long)B_SZ * IN_SZ, IN_SZ);

    // --- layer 1: [1024 x 9216] x [9216 x 2048] ---
    {
        int blocks = (B_SZ / 16) * (H1_SZ / 64) / 8;   // 256
        skan_gemm_f16_wmma<<<blocks, 256, 0, stream>>>(
            A1, Bt1, h, nullptr, scale1, nullptr, B_SZ, H1_SZ, H1_SZ, K1);
    }
    skan_bn_stats<<<H1_SZ, 256, 0, stream>>>(h, mu, vr, B_SZ, H1_SZ);
    skan_bn_spike<<<1024, 256, 0, stream>>>(h, mu, vr, gamma1, beta1,
                                            s1out, A2, (long)B_SZ * H1_SZ, H1_SZ);

    // --- layer 2 (binary-input collapse): h2 = (s1 @ dc^T + base) * scale2 ---
    {
        int blocks = (B_SZ / 16) * (H2_SZ / 64) / 8;   // 128
        skan_gemm_f16_wmma<<<blocks, 256, 0, stream>>>(
            A2, dc, h, base, scale2, nullptr, B_SZ, H2_SZ, H2_SZ, H1_SZ);
    }
    skan_bn_stats<<<H2_SZ, 256, 0, stream>>>(h, mu, vr, B_SZ, H2_SZ);
    skan_bn_spike<<<1024, 256, 0, stream>>>(h, mu, vr, gamma2, beta2,
                                            s2out, A3, (long)B_SZ * H2_SZ, H2_SZ);

    // --- classifier: out = s2 @ w^T + b  (N padded 1000 -> 1024) ---
    {
        int blocks = (B_SZ / 16) * (NC_PAD / 64) / 8;  // 128
        skan_gemm_f16_wmma<<<blocks, 256, 0, stream>>>(
            A3, wt, out, nullptr, nullptr, bvec, B_SZ, NC_PAD, NC_SZ, H2_SZ);
    }
}
